// SoftScorer_27144193311411
// MI455X (gfx1250) — compile-verified, matrix operand order checked by reading, not run on hardware
//
#include <hip/hip_runtime.h>
#include <hip/hip_bf16.h>

// ---- problem constants (match reference) ----
#define NB    8        // batch
#define TG    8192     // groundings per row
#define NM    4        // atoms per grounding
#define ND    256      // embedding dim
#define NE    2000     // num entities (hash base)
#define TOPK  1024     // output budget

typedef _Float16 v16h __attribute__((ext_vector_type(16)));
typedef float    v8f  __attribute__((ext_vector_type(8)));

struct f8 { float v[8]; };

__device__ __forceinline__ f8 load8(const float* __restrict__ p) {
    // p is 32B-aligned by construction (offset is a multiple of 8 floats)
    const float4* q = (const float4*)p;
    float4 a = q[0], b = q[1];
    f8 r;
    r.v[0] = a.x; r.v[1] = a.y; r.v[2] = a.z; r.v[3] = a.w;
    r.v[4] = b.x; r.v[5] = b.y; r.v[6] = b.z; r.v[7] = b.w;
    return r;
}

// membership in a sorted int32 table (jnp.searchsorted left + equality)
__device__ __forceinline__ bool in_sorted(int h, const int* __restrict__ t, int n) {
    int lo = 0, hi = n;
    while (lo < hi) {
        int mid = (lo + hi) >> 1;
        if (t[mid] < h) lo = mid + 1; else hi = mid;
    }
    return (lo < n) && (t[lo] == h);
}

// ---------------------------------------------------------------------------
// Kernel 1: per-grounding confidence. One wave handles a 16-atom tile
// (= 4 groundings). DistMult scores via v_wmma_f32_16x16x32_f16:
//   A[16x32] = fp16 chunks of (S ⊙ O), row i = atom i
//   B[32x16] = fp16 chunks of P,       col i = atom i
//   score_i  = diag(C)[i] accumulated over 8 chunks (D = 256)
// ---------------------------------------------------------------------------
__global__ void __launch_bounds__(256)
score_conf_kernel(const int* __restrict__ body,
                  const unsigned char* __restrict__ mask,
                  const float* __restrict__ pred_emb,
                  const float* __restrict__ ent_emb,
                  const int* __restrict__ fact_h, int n_fact,
                  const int* __restrict__ fp_h,   int n_fp,
                  float* __restrict__ conf) {
    const int lane = threadIdx.x & 31;
    const int tile = blockIdx.x * (blockDim.x >> 5) + (threadIdx.x >> 5);
    const int nTiles = (NB * TG * NM) / 16;
    if (tile >= nTiles) return;                 // uniform per-wave (exact grid)

    const int i    = lane & 15;                 // atom index within tile (= row M = col N)
    const int atom = tile * 16 + i;
    const int p = body[atom * 3 + 0];
    const int s = body[atom * 3 + 1];
    const int o = body[atom * 3 + 2];

    // warm the hash table path in GL2 (global_prefetch_b8)
    __builtin_prefetch(fact_h + (n_fact >> 1), 0, 0);
    __builtin_prefetch(fp_h + (n_fp >> 1), 0, 0);

    // hash-set membership: facts | forward-chaining fixed point
    const int h = p * (NE * NE) + s * NE + o;
    const bool known = in_sorted(h, fact_h, n_fact) || in_sorted(h, fp_h, n_fp);

    const float* __restrict__ sv = ent_emb  + (size_t)s * ND;
    const float* __restrict__ ov = ent_emb  + (size_t)o * ND;
    const float* __restrict__ pv = pred_emb + (size_t)p * ND;

    // ISA A-operand K pattern: halves j=0..7 -> K = base + g8 + j,
    //                          halves j=8..15 -> K = base + 16 + g8 + (j-8)
    const int g8 = ((lane >> 4) & 1) << 3;

    v8f c = {};
    #pragma unroll
    for (int ch = 0; ch < ND / 32; ++ch) {
        const int k0 = ch * 32 + g8;
        const int k1 = k0 + 16;
        f8 sa = load8(sv + k0), sb = load8(sv + k1);
        f8 oa = load8(ov + k0), ob = load8(ov + k1);
        f8 pa = load8(pv + k0), pb = load8(pv + k1);
        v16h A, Bm;
        #pragma unroll
        for (int j = 0; j < 8; ++j) {
            A[j]      = (_Float16)(sa.v[j] * oa.v[j]);
            A[j + 8]  = (_Float16)(sb.v[j] * ob.v[j]);
            Bm[j]     = (_Float16)pa.v[j];
            Bm[j + 8] = (_Float16)pb.v[j];
        }
        // 8 args: (neg_a, A, neg_b, B, c_mod, C, reuse_a, reuse_b)
        c = __builtin_amdgcn_wmma_f32_16x16x32_f16(false, A, false, Bm,
                                                   (short)0, c, false, false);
    }

    // Diagonal extraction from C layout (VGPR r: M=r lanes 0-15, M=8+r lanes 16-31;
    // N = lane & 15). Every lane exposes c[lane&7]; diag consumer i pulls from
    // src = i (i<8) or i+16 (i>=8), which holds exactly c[i&7] = C[i][i].
    float x = c[0];
    #pragma unroll
    for (int r = 1; r < 8; ++r) x = ((lane & 7) == r) ? c[r] : x;
    const int src = (i < 8) ? i : (i + 16);
    const float score = __shfl(x, src, 32);

    const float soft = 1.0f / (1.0f + expf(-score));
    float atom_score = (p != 0) ? (known ? 1.0f : soft) : 1.0f;

    // product over the 4 atoms of each grounding (lane quads)
    atom_score *= __shfl_xor(atom_score, 1, 32);
    atom_score *= __shfl_xor(atom_score, 2, 32);

    if (lane < 16 && (lane & 3) == 0) {
        const int g = tile * 4 + (lane >> 2);   // flat grounding index in [0, NB*TG)
        conf[g] = atom_score * (mask[g] ? 1.0f : 0.0f);
    }
}

// ---------------------------------------------------------------------------
// Kernel 2: exact per-row top-K via in-LDS bitonic sort (descending value,
// ascending index tiebreak, matching jax.lax.top_k). 64 KB of the 320 KB
// WGP LDS; 1024 threads/row.
// ---------------------------------------------------------------------------
__global__ void __launch_bounds__(1024)
topk_kernel(const float* __restrict__ conf,
            const int* __restrict__ body,
            const unsigned char* __restrict__ mask,
            const int* __restrict__ rule_idx,
            float* __restrict__ out) {
    __shared__ float sval[TG];
    __shared__ int   sidx[TG];
    const int b   = blockIdx.x;
    const int tid = threadIdx.x;
    const int nth = blockDim.x;

    for (int i = tid; i < TG; i += nth) { sval[i] = conf[b * TG + i]; sidx[i] = i; }
    __syncthreads();

    for (int k = 2; k <= TG; k <<= 1) {
        for (int j = k >> 1; j > 0; j >>= 1) {
            for (int i = tid; i < TG; i += nth) {
                const int ixj = i ^ j;
                if (ixj > i) {
                    const float vi = sval[i], vj = sval[ixj];
                    const int   ii = sidx[i], ij = sidx[ixj];
                    const bool i_first = (vi > vj) || (vi == vj && ii < ij);
                    const bool up = ((i & k) == 0);          // "up" runs: larger first
                    if (up ? !i_first : i_first) {
                        sval[i] = vj; sval[ixj] = vi;
                        sidx[i] = ij; sidx[ixj] = ii;
                    }
                }
            }
            __syncthreads();
        }
    }

    // outputs concatenated flat, value-cast to float:
    // body_sel [NB,TOPK,NM,3] | mask_sel [NB,TOPK] | rule_sel [NB,TOPK] | conf_sel [NB,TOPK]
    float* out_body = out;
    float* out_mask = out_body + (size_t)NB * TOPK * NM * 3;
    float* out_rule = out_mask + (size_t)NB * TOPK;
    float* out_conf = out_rule + (size_t)NB * TOPK;

    for (int t = tid; t < TOPK; t += nth) {
        const int idx = sidx[t];
        const int g   = b * TG + idx;
        #pragma unroll
        for (int m = 0; m < NM; ++m)
            #pragma unroll
            for (int cc = 0; cc < 3; ++cc)
                out_body[((size_t)(b * TOPK + t) * NM + m) * 3 + cc] =
                    (float)body[((size_t)g * NM + m) * 3 + cc];
        out_mask[b * TOPK + t] = mask[g] ? 1.0f : 0.0f;
        out_rule[b * TOPK + t] = (float)rule_idx[g];
        out_conf[b * TOPK + t] = sval[t];
    }
}

extern "C" void kernel_launch(void* const* d_in, const int* in_sizes, int n_in,
                              void* d_out, int out_size, void* d_ws, size_t ws_size,
                              hipStream_t stream) {
    const int*           body     = (const int*)d_in[0];
    const unsigned char* mask     = (const unsigned char*)d_in[1];
    const int*           rule_idx = (const int*)d_in[2];
    const float*         pred_emb = (const float*)d_in[3];
    const float*         ent_emb  = (const float*)d_in[4];
    const int*           fact_h   = (const int*)d_in[5];
    const int*           fp_h     = (const int*)d_in[6];
    const int n_fact = in_sizes[5];
    const int n_fp   = in_sizes[6];

    float* conf = (float*)d_ws;        // NB*TG floats = 256 KB scratch
    float* out  = (float*)d_out;

    const int nTiles = (NB * TG * NM) / 16;   // 16384 waves, 1 tile each
    const int wavesPerBlock = 8;              // 256 threads = 8 wave32s
    score_conf_kernel<<<nTiles / wavesPerBlock, wavesPerBlock * 32, 0, stream>>>(
        body, mask, pred_emb, ent_emb, fact_h, n_fact, fp_h, n_fp, conf);

    topk_kernel<<<NB, 1024, 0, stream>>>(conf, body, mask, rule_idx, out);

    (void)n_in; (void)out_size; (void)ws_size;
}
